// MSGroupConv1dSplit_37744172597310
// MI455X (gfx1250) — compile-verified, matrix operand order checked by reading, not run on hardware
//
#include <hip/hip_runtime.h>
#include <stdint.h>

// ---------------- types for WMMA ----------------
typedef __bf16 bf16_t;
typedef bf16_t v16bf __attribute__((ext_vector_type(16)));
typedef float  v8f   __attribute__((ext_vector_type(8)));

union ABFrag { unsigned u[8]; uint4 q[2]; v16bf v; };

#define NT   64
#define CC   256
#define CQ   64
#define HH_  56
#define WW_  56
#define HW   3136
#define TSEG 8
#define FOLD 8

// ybf: [p][n][pix][cin] bf16  -> 4*64*3136*64 elems
#define YBF_BYTES ((size_t)4 * NT * HW * CQ * 2)   // 102,760,448 B
// wpk: [p][tap][kb][ntile][half][m][r] dwords -> 4*9*2*4*2*16*8 = 73728
#define WPK_DWORDS (4 * 9 * 2048)

__device__ __forceinline__ unsigned short f2bf(float f) {
    unsigned u = __float_as_uint(f);
    u += 0x7FFFu + ((u >> 16) & 1u);   // round to nearest even
    return (unsigned short)(u >> 16);
}

// ---------------- phase 1: temporal shift + bf16 NHWC staging ----------------
__global__ __launch_bounds__(256)
void shift_bf16_kernel(const float* __restrict__ x, unsigned short* __restrict__ ybf) {
    __shared__ unsigned short tile[64][66];          // [c][pix], padded
    const int pixBase = blockIdx.x * 64;             // 49 blocks (3136/64)
    const int n = blockIdx.y;                        // 0..63
    const int p = blockIdx.z;                        // 0..3
    const int t = n & (TSEG - 1);
    const int tid = threadIdx.x;

    #pragma unroll
    for (int i = 0; i < 16; ++i) {
        int e   = tid + i * 256;                     // 4096 elems
        int c   = e >> 6;
        int pix = e & 63;
        int dtlo = 0, dthi = 0;
        if (p > 0) {
            if (c < FOLD)          { dtlo = -p; dthi = -1; }   // backward window sum
            else if (c < 2 * FOLD) { dtlo =  1; dthi =  p; }   // forward window sum
        }
        float v = 0.f;
        for (int dt = dtlo; dt <= dthi; ++dt) {
            int tt = t + dt;
            if (0 <= tt && tt < TSEG) {
                size_t idx = ((size_t)(n - t + tt) * CC + (p * CQ + c)) * HW + pixBase + pix;
                v += x[idx];
            }
        }
        tile[c][pix] = f2bf(v);
    }
    __syncthreads();

    // transposed, dword-packed coalesced write: [pix][c]
    unsigned* yb = (unsigned*)ybf;
    size_t base = (((size_t)p * NT + n) * HW + pixBase) * (CQ / 2);
    #pragma unroll
    for (int i = 0; i < 8; ++i) {
        int e   = tid + i * 256;                     // 2048 dwords
        int pix = e >> 5;
        int cp  = e & 31;
        unsigned lo = tile[2 * cp][pix];
        unsigned hi = tile[2 * cp + 1][pix];
        yb[base + (size_t)pix * (CQ / 2) + cp] = lo | (hi << 16);
    }
}

// ---------------- phase 1b: pack weights into exact B-fragment order ----------
// layout: [p][tap][kb][ntile][half][m][r] ; dword = bf16 pair for cins (2kp,2kp+1)
// with kp = ((r<4)?r:r+4) + 4*half + 16*kb, cout = 16*ntile + m.
__global__ __launch_bounds__(256)
void wpack_kernel(const float* __restrict__ w0, const float* __restrict__ w1,
                  const float* __restrict__ w2, const float* __restrict__ w3,
                  unsigned* __restrict__ wpk) {
    int id = blockIdx.x * 256 + threadIdx.x;
    if (id >= WPK_DWORDS) return;
    int r    = id & 7;
    int m    = (id >> 3) & 15;
    int half = (id >> 7) & 1;
    int t    = (id >> 8) & 3;
    int kb   = (id >> 10) & 1;
    int r2   = id >> 11;             // p*9 + tap
    int tap  = r2 % 9;
    int p    = r2 / 9;
    const float* ws = (p == 0) ? w0 : (p == 1) ? w1 : (p == 2) ? w2 : w3;
    int kp   = ((r < 4) ? r : r + 4) + 4 * half + 16 * kb;
    int cout = 16 * t + m;
    int cin  = 2 * kp;
    unsigned lo = f2bf(ws[((size_t)cout * CQ + cin) * 9 + tap]);
    unsigned hi = f2bf(ws[((size_t)cout * CQ + cin + 1) * 9 + tap]);
    wpk[id] = lo | (hi << 16);
}

// ---------------- phase 2: implicit-GEMM conv via WMMA bf16 ----------------
// wave task: (p, n, group of 2 M-tiles). 2 Mx16 pixel tiles x 4 Nx16 cout
// tiles, K = 9 taps * 64 cin (2 k-blocks of 32) -> 144 v_wmma per wave.
// All fragments loaded as 2x global_load_b128.
__global__ __launch_bounds__(256)
void conv_wmma_kernel(const unsigned* __restrict__ ybfU,
                      const unsigned* __restrict__ wpkU,
                      float* __restrict__ out) {
    __shared__ __align__(16) float sout[8][64 * 20];   // per-wave staging, padded rows

    const int lane = threadIdx.x & 31;
    const int warp = threadIdx.x >> 5;
    const int wid  = blockIdx.x * 8 + warp;            // 25088 waves total
    const int tg   = wid % 98;                         // 98 tile-pairs per image
    const int n    = (wid / 98) & 63;
    const int p    = wid / (98 * 64);
    const int m    = lane & 15;                        // A-row / B-col index
    const int half = lane >> 4;

    int hh[2], ww[2];
    #pragma unroll
    for (int mt = 0; mt < 2; ++mt) {
        int pix = (tg * 2 + mt) * 16 + m;
        hh[mt] = pix / WW_;
        ww[mt] = pix % WW_;
    }

    v8f acc[2][4];
    #pragma unroll
    for (int mt = 0; mt < 2; ++mt)
        #pragma unroll
        for (int t = 0; t < 4; ++t)
            #pragma unroll
            for (int j = 0; j < 8; ++j) acc[mt][t][j] = 0.f;

    const size_t ybase = (((size_t)p * NT + n) * HW) * 32;       // 32 dwords / pixel
    const unsigned* wp = wpkU + (size_t)p * 9 * 2048;
    const int laneOffB = (half * 16 + m) * 8;                    // dwords into fragment block

    for (int kh = -1; kh <= 1; ++kh) {
        for (int kw = -1; kw <= 1; ++kw) {
            const int tap = (kh + 1) * 3 + (kw + 1);
            const unsigned* wtap = wp + tap * 2048;
            #pragma unroll
            for (int kb = 0; kb < 2; ++kb) {
                ABFrag bfr[4];
                #pragma unroll
                for (int t = 0; t < 4; ++t) {
                    const uint4* bq = (const uint4*)(wtap + ((kb * 4 + t) * 2) * 128 + laneOffB);
                    bfr[t].q[0] = bq[0];
                    bfr[t].q[1] = bq[1];
                }
                #pragma unroll
                for (int mt = 0; mt < 2; ++mt) {
                    int ih = hh[mt] + kh, iw = ww[mt] + kw;
                    bool valid = ((unsigned)ih < (unsigned)HH_) && ((unsigned)iw < (unsigned)WW_);
                    ABFrag a;
                    if (valid) {
                        const uint4* aq = (const uint4*)(ybfU + ybase +
                            (size_t)(ih * WW_ + iw) * 32 + kb * 16 + half * 4);
                        a.q[0] = aq[0];
                        a.q[1] = aq[2];   // +8 dwords
                    } else {
                        #pragma unroll
                        for (int r = 0; r < 8; ++r) a.u[r] = 0u;
                    }
                    #pragma unroll
                    for (int t = 0; t < 4; ++t)
                        acc[mt][t] = __builtin_amdgcn_wmma_f32_16x16x32_bf16(
                            false, a.v, false, bfr[t].v, (short)0, acc[mt][t], false, false);
                }
            }
        }
    }

    // D layout: reg j, lane -> M = j + 8*half, N = lane%16. Transpose through
    // per-wave LDS, then coalesced b128 stores into NCHW output.
    float* slds = sout[warp];
    const size_t outn = (size_t)n * CC * HW + (size_t)p * CQ * HW;
    const int cl = lane >> 2;            // 0..7
    const int p4 = (lane & 3) * 4;       // 0,4,8,12
    #pragma unroll
    for (int mt = 0; mt < 2; ++mt) {
        #pragma unroll
        for (int t = 0; t < 4; ++t)
            #pragma unroll
            for (int j = 0; j < 8; ++j)
                slds[(t * 16 + m) * 20 + j + 8 * half] = acc[mt][t][j];
        asm volatile("s_wait_dscnt 0" ::: "memory");   // same-wave cross-lane RAW
        int pixb = (tg * 2 + mt) * 16;
        #pragma unroll
        for (int c8 = 0; c8 < 8; ++c8) {
            int cout = c8 * 8 + cl;
            float4 v = *(const float4*)&slds[cout * 20 + p4];
            *(float4*)&out[outn + (size_t)cout * HW + pixb + p4] = v;
        }
        asm volatile("s_wait_dscnt 0" ::: "memory");   // drain reads before next mt overwrite
    }
}

// ---------------- host launcher ----------------
extern "C" void kernel_launch(void* const* d_in, const int* in_sizes, int n_in,
                              void* d_out, int out_size, void* d_ws, size_t ws_size,
                              hipStream_t stream) {
    const float* x   = (const float*)d_in[0];
    // d_in[1..4] = w1,w3,w5,w7 (fixed binary shift masks; semantics hardcoded)
    const float* w21 = (const float*)d_in[5];
    const float* w22 = (const float*)d_in[6];
    const float* w23 = (const float*)d_in[7];
    const float* w24 = (const float*)d_in[8];

    unsigned short* ybf = (unsigned short*)d_ws;
    unsigned*       wpk = (unsigned*)((char*)d_ws + YBF_BYTES);
    float*          out = (float*)d_out;

    wpack_kernel<<<(WPK_DWORDS + 255) / 256, 256, 0, stream>>>(w21, w22, w23, w24, wpk);
    shift_bf16_kernel<<<dim3(49, 64, 4), 256, 0, stream>>>(x, ybf);
    conv_wmma_kernel<<<3136, 256, 0, stream>>>((const unsigned*)ybf, wpk, out);
}
